// InpaintGeneratorFull_75866302317133
// MI455X (gfx1250) — matrix operand order repaired and checked by reading, not compiled, not run on hardware
//
#include <hip/hip_runtime.h>
#include <hip/hip_bf16.h>
#include <math.h>

// ---------------------------------------------------------------------------
// CDNA5 (gfx1250) wave32 WMMA types
// ---------------------------------------------------------------------------
typedef __attribute__((ext_vector_type(16))) _Float16 v16h;
typedef __attribute__((ext_vector_type(8)))  float    v8f;

// A-fragment K mapping for V_WMMA_F32_16X16X32_F16 (16-bit A 16x32):
// lanes 0-15: regs0-3 -> K=0..7, regs4-7 -> K=16..23
// lanes16-31: regs0-3 -> K=8..15, regs4-7 -> K=24..31
__device__ __forceinline__ int a_kmap(int e, int hf) {
    return (e < 8) ? (hf * 8 + e) : (16 + hf * 8 + (e - 8));
}

// _unfold2 patch element:  p = ((c*4 + quad)*s + i)*s + j ; patch l = lh*nw+lw
// value = img_padded[b,c,(qh+lh)*s+i, (qw+lw)*s+j] with pd zero-padding.
// Branchless: clamp OOB address to a valid location, select 0 on the value —
// avoids EXEC-masked predicated loads (s_and_saveexec/s_cbranch per element).
__device__ __forceinline__ float patch_val(const float* img, int b, int C, int HW,
                                           int l, int p, int s, int pd, int nw) {
    int ss = s * s;
    int c    = p / (4 * ss);
    int r    = p % (4 * ss);
    int quad = r / ss;
    int ij   = r % ss;
    int i = ij / s, j = ij % s;
    int qh = quad >> 1, qw = quad & 1;
    int lh = l / nw,   lw = l % nw;
    int y = (qh + lh) * s + i - pd;
    int x = (qw + lw) * s + j - pd;
    bool ok = (y >= 0) && (y < HW) && (x >= 0) && (x < HW);
    int yc = ok ? y : 0;
    int xc = ok ? x : 0;
    float v = img[((size_t)(b * C + c) * HW + yc) * HW + xc];
    return ok ? v : 0.0f;
}

// ---------------------------------------------------------------------------
// Implicit-GEMM convolution, fully templated shape so all index div/mod
// lowers to mul-shift (no runtime integer-division chains feeding the WMMAs).
// 16x16 output tile per wave, 4 waves/block, K-steps of 32 (f16 -> f32 acc).
// M = B*HO*WO (output pixels), N = COUT, K = CIN*KK*KK (zero-padded to 32).
// ---------------------------------------------------------------------------
template <int CIN, int HIN, int COUT, int KK, int STRIDE>
__global__ void k_conv_wmma(const float* __restrict__ in, const float* __restrict__ wgt,
                            const float* __restrict__ bias, float* __restrict__ out,
                            int B) {
    constexpr int PAD  = (KK - 1) / 2;
    constexpr int HO   = (HIN + 2 * PAD - KK) / STRIDE + 1;
    constexpr int WO   = HO;
    constexpr int KDIM = CIN * KK * KK;
    constexpr int KKQ  = KK * KK;

    const int lane  = threadIdx.x & 31;
    const int wv    = threadIdx.x >> 5;
    const int tileM = blockIdx.x * 4 + wv;
    const int tileN = blockIdx.y;
    const int M     = B * HO * WO;
    if (tileM * 16 >= M) return;                 // wave-uniform: EXEC stays all-1s
    const int hf = lane >> 4;

    const int mrow = tileM * 16 + (lane & 15);   // A row for this lane
    int bb  = mrow / (HO * WO);
    int rem = mrow % (HO * WO);
    int oy  = rem / WO, ox = rem % WO;
    int iy0 = oy * STRIDE - PAD;
    int ix0 = ox * STRIDE - PAD;
    const float* inb = in + (size_t)bb * CIN * HIN * HIN;

    v8f acc = {};
    for (int kk = 0; kk < KDIM; kk += 32) {
        v16h a, bfr;
#pragma unroll
        for (int e = 0; e < 16; ++e) {
            int kidx = kk + a_kmap(e, hf);
            int ci = kidx / KKQ;
            int r  = kidx % KKQ;
            int ky = r / KK, kx = r % KK;
            int iy = iy0 + ky, ix = ix0 + kx;
            bool ok = (KDIM % 32 == 0 || kidx < KDIM) &&
                      iy >= 0 && iy < HIN && ix >= 0 && ix < HIN;
            int cic = ok ? ci : 0;
            int iyc = ok ? iy : 0;
            int ixc = ok ? ix : 0;
            float v = inb[((size_t)cic * HIN + iyc) * HIN + ixc];
            a[e] = (_Float16)(ok ? v : 0.0f);
        }
        int  kb  = kk + lane;                    // lane = K offset, used only by the prefetch
        bool kok = (KDIM % 32 == 0) || (kb < KDIM);
#pragma unroll
        for (int e = 0; e < 16; ++e) {           // B: lane%16 = N (out channel), element = K (a_kmap)
            int  kbe = kk + a_kmap(e, hf);
            int  co = tileN * 16 + (lane & 15);
            bool ok = (KDIM % 32 == 0 || kbe < KDIM) && (COUT % 16 == 0 || co < COUT);
            size_t idx = ok ? ((size_t)co * KDIM + kbe) : 0;
            float  v   = wgt[idx];
            bfr[e] = (_Float16)(ok ? v : 0.0f);
        }
        if (kk + 32 < KDIM)                      // global_prefetch_b8 of next K-step
            __builtin_prefetch(&wgt[(size_t)(tileN * 16) * KDIM + kb + 32], 0, 1);
        acc = __builtin_amdgcn_wmma_f32_16x16x32_f16(false, a, false, bfr,
                                                     (short)0, acc, false, false);
    }
    // C/D: lane%16 = N, reg r -> M = r + 8*(lane/16)
    int co = tileN * 16 + (lane & 15);
    if (COUT % 16 == 0 || co < COUT) {
        float bval = bias[co];
#pragma unroll
        for (int r = 0; r < 8; ++r) {
            int m    = tileM * 16 + r + 8 * hf;
            int bb2  = m / (HO * WO);
            int rem2 = m % (HO * WO);
            int oy2  = rem2 / WO, ox2 = rem2 % WO;
            out[((size_t)(bb2 * COUT + co) * HO + oy2) * WO + ox2] = acc[r] + bval;
        }
    }
}

// ---------------------------------------------------------------------------
// cp1: cos[b,l,q] = (sum_p patch(l,p)*rnorm[l] * patch(q,p)) * mm[l]
// M=l, N=q, K=p=1536. grid(16, 64, 4), block 128.
// ---------------------------------------------------------------------------
__global__ void k_cp1_cos_wmma(const float* __restrict__ xsim, const float* __restrict__ rnorm,
                               const float* __restrict__ mm, float* __restrict__ cosb) {
    const int lane = threadIdx.x & 31;
    const int wv   = threadIdx.x >> 5;
    const int tileM = blockIdx.x * 4 + wv;
    const int tileN = blockIdx.y;
    const int b     = blockIdx.z;
    const int hf    = lane >> 4;
    const int L = 1024, P = 1536, C = 96, HW = 64, s = 2, pd = 1, nw = 32;

    int   lrow = tileM * 16 + (lane & 15);
    float rn   = rnorm[b * L + lrow];
    v8f acc = {};
    for (int kk = 0; kk < P; kk += 32) {
        v16h a, bfr;
#pragma unroll
        for (int e = 0; e < 16; ++e) {
            int p = kk + a_kmap(e, hf);
            a[e] = (_Float16)(patch_val(xsim, b, C, HW, lrow, p, s, pd, nw) * rn);
        }
        int q = tileN * 16 + (lane & 15);
#pragma unroll
        for (int e = 0; e < 16; ++e) {
            int pb = kk + a_kmap(e, hf);
            bfr[e] = (_Float16)patch_val(xsim, b, C, HW, q, pb, s, pd, nw);
        }
        acc = __builtin_amdgcn_wmma_f32_16x16x32_f16(false, a, false, bfr,
                                                     (short)0, acc, false, false);
    }
    int q = tileN * 16 + (lane & 15);
#pragma unroll
    for (int r = 0; r < 8; ++r) {
        int l = tileM * 16 + r + 8 * hf;
        cosb[((size_t)b * L + l) * L + q] = acc[r] * mm[b * L + l];
    }
}

// ---------------------------------------------------------------------------
// cp2: contrib[b,q,p] = sum_l sim[b,l,q] * patch(xb,l,p); fold fused as
// atomic scatter-add into padded (B,24,264,264) buffer.
// M=q, N=p(6144), K=l=1024. grid(16, 384, 4), block 128.
// ---------------------------------------------------------------------------
__global__ void k_cp2_wmma(const float* __restrict__ sim, const float* __restrict__ xb,
                           float* __restrict__ outpad) {
    const int lane = threadIdx.x & 31;
    const int wv   = threadIdx.x >> 5;
    const int tileM = blockIdx.x * 4 + wv;
    const int tileN = blockIdx.y;
    const int b     = blockIdx.z;
    const int hf    = lane >> 4;
    const int L = 1024, C = 24, HW = 256, s = 8, pd = 4, nw = 32, Hp = 264;

    int qrow = tileM * 16 + (lane & 15);
    v8f acc = {};
    for (int kk = 0; kk < L; kk += 32) {
        v16h a, bfr;
#pragma unroll
        for (int e = 0; e < 16; ++e) {           // A[q][l] = sim[b][l][q]
            int l = kk + a_kmap(e, hf);
            a[e] = (_Float16)sim[((size_t)b * L + l) * L + qrow];
        }
        int p = tileN * 16 + (lane & 15);
#pragma unroll
        for (int e = 0; e < 16; ++e) {           // B[l][p] = patch(xb, l, p)
            int lb = kk + a_kmap(e, hf);
            bfr[e] = (_Float16)patch_val(xb, b, C, HW, lb, p, s, pd, nw);
        }
        if (kk + 32 < L)
            __builtin_prefetch(&sim[((size_t)b * L + kk + 32) * L + qrow], 0, 1);
        acc = __builtin_amdgcn_wmma_f32_16x16x32_f16(false, a, false, bfr,
                                                     (short)0, acc, false, false);
    }
    // fold: element (q,p) -> unique padded pixel; overlaps resolved by atomics
    int p    = tileN * 16 + (lane & 15);
    int c    = p / 256;
    int r0   = p % 256;
    int quad = r0 / 64;
    int ij   = r0 % 64;
    int i = ij / 8, j = ij % 8;
    int qh = quad >> 1, qw = quad & 1;
#pragma unroll
    for (int r = 0; r < 8; ++r) {
        int q  = tileM * 16 + r + 8 * hf;
        int lh = q / nw, lw = q % nw;
        int y = (qh + lh) * s + i;
        int x = (qw + lw) * s + j;
        atomicAdd(&outpad[((size_t)(b * C + c) * Hp + y) * Hp + x], acc[r]);
    }
}

// ---------------------------------------------------------------------------
// Elementwise / reduction helpers
// ---------------------------------------------------------------------------
__global__ void k_zero(float* p, int n) {
    int i = blockIdx.x * blockDim.x + threadIdx.x;
    if (i < n) p[i] = 0.0f;
}

__global__ void k_blend_xnow(const float* __restrict__ x, const float* __restrict__ xs2,
                             const float* __restrict__ mask, float* __restrict__ xnow, int n) {
    int i = blockIdx.x * blockDim.x + threadIdx.x;
    if (i >= n) return;
    int b = i / (3 * 65536), pix = i % 65536;
    float m = mask[b * 65536 + pix];
    xnow[i] = xs2[i] * m + x[i] * (1.0f - m);
}

__global__ void k_avgpool4(const float* __restrict__ mask, float* __restrict__ masks) {
    int i = blockIdx.x * blockDim.x + threadIdx.x;
    if (i >= 4 * 64 * 64) return;
    int b = i / 4096, r = i % 4096, y = r / 64, x = r % 64;
    float s = 0.0f;
    for (int dy = 0; dy < 4; ++dy)
        for (int dx = 0; dx < 4; ++dx)
            s += mask[(size_t)b * 65536 + (y * 4 + dy) * 256 + (x * 4 + dx)];
    masks[i] = s * (1.0f / 16.0f);
}

// gated activation: out = act(y[:,0:C]) * sigmoid(y[:,C:2C]); act 0=elu,1=relu
__global__ void k_gate(const float* __restrict__ raw, float* __restrict__ out,
                       int B, int Ch, int HW, int act) {
    int i = blockIdx.x * blockDim.x + threadIdx.x;
    int n = B * Ch * HW;
    if (i >= n) return;
    int b = i / (Ch * HW), r = i % (Ch * HW);
    float a = raw[(size_t)b * (2 * Ch) * HW + r];
    float g = raw[(size_t)b * (2 * Ch) * HW + (size_t)Ch * HW + r];
    float av = (act == 1) ? fmaxf(a, 0.0f) : (a > 0.0f ? a : __expf(a) - 1.0f);
    out[i] = av * (1.0f / (1.0f + __expf(-g)));
}

__global__ void k_up2(const float* __restrict__ in, float* __restrict__ out, int BC, int H) {
    int H2 = 2 * H;
    int n  = BC * H2 * H2;
    int i  = blockIdx.x * blockDim.x + threadIdx.x;
    if (i >= n) return;
    int bc = i / (H2 * H2), r = i % (H2 * H2), y = r / H2, x = r % H2;
    out[i] = in[(size_t)bc * H * H + (y >> 1) * H + (x >> 1)];
}

__global__ void k_concat(const float* __restrict__ a, const float* __restrict__ b,
                         float* __restrict__ out, int B, int C1, int C2, int HW) {
    int Ct = C1 + C2;
    int n  = B * Ct * HW;
    int i  = blockIdx.x * blockDim.x + threadIdx.x;
    if (i >= n) return;
    int bb = i / (Ct * HW), r = i % (Ct * HW), c = r / HW, pix = r % HW;
    out[i] = (c < C1) ? a[((size_t)bb * C1 + c) * HW + pix]
                      : b[((size_t)bb * C2 + (c - C1)) * HW + pix];
}

// per-(b,l): 1/max(||patch||,1e-4) over 1536 elems; mm = (mask patch sum == 0)
__global__ void k_patchstat(const float* __restrict__ xsim, const float* __restrict__ masks,
                            float* __restrict__ rnorm, float* __restrict__ mm) {
    int bl = blockIdx.x;
    int b  = bl >> 10, l = bl & 1023;
    int lane = threadIdx.x;
    float ss = 0.0f;
    for (int p = lane; p < 1536; p += 32) {
        float v = patch_val(xsim, b, 96, 64, l, p, 2, 1, 32);
        ss += v * v;
    }
    for (int off = 16; off > 0; off >>= 1) ss += __shfl_xor(ss, off, 32);
    if (lane == 0) {
        rnorm[bl] = 1.0f / fmaxf(sqrtf(ss), 1e-4f);
        float msum = 0.0f;
        for (int p = 0; p < 16; ++p)
            msum += patch_val(masks, b, 1, 64, l, p, 2, 1, 32);
        mm[bl] = (msum == 0.0f) ? 1.0f : 0.0f;
    }
}

// softmax over l (column, stride L) of cos*10, then * mm[l]; in-place
__global__ void k_softmax_col(float* __restrict__ cosb, const float* __restrict__ mm) {
    const int L = 1024;
    int b = blockIdx.x >> 10, q = blockIdx.x & 1023;
    int lane = threadIdx.x;
    float* col = cosb + (size_t)b * L * L + q;
    const float scale = 10.0f;
    float mx = -1e30f;
    for (int l = lane; l < L; l += 32) mx = fmaxf(mx, col[(size_t)l * L] * scale);
    for (int off = 16; off > 0; off >>= 1) mx = fmaxf(mx, __shfl_xor(mx, off, 32));
    float sum = 0.0f;
    for (int l = lane; l < L; l += 32) sum += __expf(col[(size_t)l * L] * scale - mx);
    for (int off = 16; off > 0; off >>= 1) sum += __shfl_xor(sum, off, 32);
    float inv = 1.0f / sum;
    for (int l = lane; l < L; l += 32) {
        float e = __expf(col[(size_t)l * L] * scale - mx);
        col[(size_t)l * L] = e * inv * mm[b * L + l];
    }
}

// divide by analytic fold count, crop pd=4, blend with mask
__global__ void k_cp2_finish(const float* __restrict__ outpad, const float* __restrict__ xb,
                             const float* __restrict__ mask, float* __restrict__ out) {
    const int C = 24, HW = 256, Hp = 264;
    int n = 4 * C * HW * HW;
    int i = blockIdx.x * blockDim.x + threadIdx.x;
    if (i >= n) return;
    int b = i / (C * HW * HW), r = i % (C * HW * HW), c = r / (HW * HW), pr = r % (HW * HW);
    int y = pr / HW, x = pr % HW;
    int yp = y + 4, xp = x + 4;
    int ty = yp / 8, tx = xp / 8;
    int cy = (ty == 0 || ty == 32) ? 1 : 2;
    int cx = (tx == 0 || tx == 32) ? 1 : 2;
    float v = outpad[((size_t)(b * C + c) * Hp + yp) * Hp + xp] / (float)(cy * cx);
    float m = mask[(size_t)b * 65536 + y * 256 + x];
    out[i] = v * m + xb[i] * (1.0f - m);
}

__global__ void k_tanh(const float* __restrict__ in, float* __restrict__ out, int n) {
    int i = blockIdx.x * blockDim.x + threadIdx.x;
    if (i < n) out[i] = tanhf(in[i]);
}

// ---------------------------------------------------------------------------
// Host orchestration
// ---------------------------------------------------------------------------
template <int CIN, int HIN, int COUT, int KK, int STRIDE>
static inline void launch_conv(hipStream_t st, const float* in, const float* w,
                               const float* b, float* out) {
    constexpr int PAD = (KK - 1) / 2;
    constexpr int HO  = (HIN + 2 * PAD - KK) / STRIDE + 1;
    int M = 4 * HO * HO;
    dim3 grid((M / 16 + 3) / 4, (COUT + 15) / 16, 1);
    k_conv_wmma<CIN, HIN, COUT, KK, STRIDE><<<grid, 128, 0, st>>>(in, w, b, out, 4);
}
static inline dim3 ew(int n) { return dim3((n + 255) / 256); }

extern "C" void kernel_launch(void* const* d_in, const int* in_sizes, int n_in,
                              void* d_out, int out_size, void* d_ws, size_t ws_size,
                              hipStream_t stream) {
    (void)in_sizes; (void)n_in; (void)out_size; (void)ws_size;
    const float* x     = (const float*)d_in[0];
    const float* mask  = (const float*)d_in[1];
    const float* xs1   = (const float*)d_in[2];
    const float* xs2   = (const float*)d_in[3];
    const float* pm    = (const float*)d_in[4];
    const float* w_sconv1  = (const float*)d_in[5];  const float* b_sconv1  = (const float*)d_in[6];
    const float* w_sconv2  = (const float*)d_in[7];  const float* b_sconv2  = (const float*)d_in[8];
    const float* w_bconv1  = (const float*)d_in[9];  const float* b_bconv1  = (const float*)d_in[10];
    const float* w_bconv2  = (const float*)d_in[11]; const float* b_bconv2  = (const float*)d_in[12];
    const float* w_bconv3  = (const float*)d_in[13]; const float* b_bconv3  = (const float*)d_in[14];
    const float* w_bconv4  = (const float*)d_in[15]; const float* b_bconv4  = (const float*)d_in[16];
    const float* w_conv13  = (const float*)d_in[17]; const float* b_conv13  = (const float*)d_in[18];
    const float* w_conv14  = (const float*)d_in[19]; const float* b_conv14  = (const float*)d_in[20];
    const float* w_conv15  = (const float*)d_in[21]; const float* b_conv15  = (const float*)d_in[22];
    const float* w_conv16  = (const float*)d_in[23]; const float* b_conv16  = (const float*)d_in[24];
    const float* w_conv16_2= (const float*)d_in[25]; const float* b_conv16_2= (const float*)d_in[26];
    const float* w_conv17  = (const float*)d_in[27]; const float* b_conv17  = (const float*)d_in[28];
    float* out = (float*)d_out;
    float* ws  = (float*)d_ws;

    const size_t IMG = 786432;                    // 4*3*256*256
    // workspace map (floats); total ~60.5M floats (~242 MB)
    float* XNOW   = ws;                           // 786432
    float* XSIM   = XNOW   + 786432;              // 4*96*64*64   = 1572864
    float* MASKS  = XSIM   + 1572864;             // 4*64*64      = 16384
    float* RNORM  = MASKS  + 16384;               // 4096
    float* MM     = RNORM  + 4096;                // 4096
    float* COS    = MM     + 4096;                // 4*1024*1024  = 4194304
    float* SKIP1  = COS    + 4194304;             // 4*24*256*256 = 6291456
    float* SKIP2  = SKIP1  + 6291456;             // 4*48*128*128 = 3145728
    float* OUTPAD = SKIP2  + 3145728;             // 4*24*264*264 = 6690816
    float* TMP0   = OUTPAD + 6690816;             // 12582912
    float* TMP1   = TMP0   + 12582912;            // 12582912
    float* TMP2   = TMP1   + 12582912;            // 12582912

    // outputs 0,1 are pass-through copies of x_stage1 / x_stage2
    hipMemcpyAsync(out,       xs1, IMG * sizeof(float), hipMemcpyDeviceToDevice, stream);
    hipMemcpyAsync(out + IMG, xs2, IMG * sizeof(float), hipMemcpyDeviceToDevice, stream);

    // xnow = x_stage2*mask + x*(1-mask)
    k_blend_xnow<<<ew(4 * 3 * 65536), 256, 0, stream>>>(x, xs2, mask, XNOW, 4 * 3 * 65536);
    // mask_s = 4x4 avgpool
    k_avgpool4<<<ew(4 * 64 * 64), 256, 0, stream>>>(mask, MASKS);

    // similarity branch: sconv1 (elu-gated), sconv2 (relu-gated)
    launch_conv<96, 64, 192, 5, 1>(stream, pm,   w_sconv1, b_sconv1, TMP0);
    k_gate<<<ew(4 * 96 * 4096), 256, 0, stream>>>(TMP0, TMP1, 4, 96, 4096, 0);
    launch_conv<96, 64, 192, 3, 1>(stream, TMP1, w_sconv2, b_sconv2, TMP0);
    k_gate<<<ew(4 * 96 * 4096), 256, 0, stream>>>(TMP0, XSIM, 4, 96, 4096, 1);

    // cp1: patch norms + mask flags, cosine GEMM (WMMA), column softmax
    k_patchstat<<<4096, 32, 0, stream>>>(XSIM, MASKS, RNORM, MM);
    k_cp1_cos_wmma<<<dim3(16, 64, 4), 128, 0, stream>>>(XSIM, RNORM, MM, COS);
    k_softmax_col<<<4096, 32, 0, stream>>>(COS, MM);

    // branch convs (all elu-gated)
    launch_conv<3, 256, 48, 5, 1>(stream, XNOW, w_bconv1, b_bconv1, TMP0);
    k_gate<<<ew(4 * 24 * 65536), 256, 0, stream>>>(TMP0, SKIP1, 4, 24, 65536, 0);
    launch_conv<24, 256, 96, 3, 2>(stream, SKIP1, w_bconv2, b_bconv2, TMP0);
    k_gate<<<ew(4 * 48 * 16384), 256, 0, stream>>>(TMP0, TMP1, 4, 48, 16384, 0);
    launch_conv<48, 128, 96, 3, 1>(stream, TMP1, w_bconv3, b_bconv3, TMP0);
    k_gate<<<ew(4 * 48 * 16384), 256, 0, stream>>>(TMP0, SKIP2, 4, 48, 16384, 0);
    launch_conv<48, 128, 192, 3, 2>(stream, SKIP2, w_bconv4, b_bconv4, TMP0);
    k_gate<<<ew(4 * 96 * 4096), 256, 0, stream>>>(TMP0, TMP1, 4, 96, 4096, 0);

    // conv13 deconv: up2 then conv
    k_up2<<<ew(4 * 96 * 16384), 256, 0, stream>>>(TMP1, TMP2, 4 * 96, 64);
    launch_conv<96, 128, 96, 3, 1>(stream, TMP2, w_conv13, b_conv13, TMP0);
    k_gate<<<ew(4 * 48 * 16384), 256, 0, stream>>>(TMP0, TMP1, 4, 48, 16384, 0);

    // concat + conv14
    k_concat<<<ew(4 * 96 * 16384), 256, 0, stream>>>(TMP1, SKIP2, TMP2, 4, 48, 48, 16384);
    launch_conv<96, 128, 96, 3, 1>(stream, TMP2, w_conv14, b_conv14, TMP0);
    k_gate<<<ew(4 * 48 * 16384), 256, 0, stream>>>(TMP0, TMP1, 4, 48, 16384, 0);

    // conv15 deconv
    k_up2<<<ew(4 * 48 * 65536), 256, 0, stream>>>(TMP1, TMP2, 4 * 48, 128);
    launch_conv<48, 256, 48, 3, 1>(stream, TMP2, w_conv15, b_conv15, TMP0);
    k_gate<<<ew(4 * 24 * 65536), 256, 0, stream>>>(TMP0, TMP1, 4, 24, 65536, 0);

    // concat + conv16  -> xb (TMP1)
    k_concat<<<ew(4 * 48 * 65536), 256, 0, stream>>>(TMP1, SKIP1, TMP2, 4, 24, 24, 65536);
    launch_conv<48, 256, 48, 3, 1>(stream, TMP2, w_conv16, b_conv16, TMP0);
    k_gate<<<ew(4 * 24 * 65536), 256, 0, stream>>>(TMP0, TMP1, 4, 24, 65536, 0);

    // cp2: GEMM with fused fold (atomic scatter), then normalize/crop/blend
    k_zero<<<ew(6690816), 256, 0, stream>>>(OUTPAD, 6690816);
    k_cp2_wmma<<<dim3(16, 384, 4), 128, 0, stream>>>(COS, TMP1, OUTPAD);
    k_cp2_finish<<<ew(4 * 24 * 65536), 256, 0, stream>>>(OUTPAD, TMP1, mask, TMP2);

    // conv16_2 (elu-gated), conv17 (no gate), tanh -> output slot 2
    launch_conv<24, 256, 48, 3, 1>(stream, TMP2, w_conv16_2, b_conv16_2, TMP0);
    k_gate<<<ew(4 * 24 * 65536), 256, 0, stream>>>(TMP0, TMP1, 4, 24, 65536, 0);
    launch_conv<24, 256, 3, 3, 1>(stream, TMP1, w_conv17, b_conv17, TMP0);
    k_tanh<<<ew((int)IMG), 256, 0, stream>>>(TMP0, out + 2 * IMG, (int)IMG);
}
